// VFXNetPatchDecoder_71511205479081
// MI455X (gfx1250) — matrix operand order repaired and checked by reading, not compiled, not run on hardware
//
#include <hip/hip_runtime.h>
#include <math.h>

// ---------------------------------------------------------------------------
// VFXNetPatchDecoder for MI455X (gfx1250, wave32, WMMA)
//   latent[1024,1024,4] f32 gather (L2-resident) -> 16x108 patch (f16 in LDS)
//   -> 4-layer MLP via v_wmma_f32_16x16x32_f16, f32 accum, GELU/sigmoid.
// ---------------------------------------------------------------------------

typedef __attribute__((ext_vector_type(16))) _Float16 v16h;
typedef __attribute__((ext_vector_type(8)))  _Float16 v8h;
typedef __attribute__((ext_vector_type(4)))  _Float16 v4h;
typedef __attribute__((ext_vector_type(8)))  float    v8f;

#define HW   1024
#define NPIX (HW * HW)

// ws layout (halfs): 50 fragments x 512 halfs (32 lanes x 16 halfs each)
//   L1: frags  0..15 (K=128pad, N=64 : 4 kT x 4 nT)
//   L2: frags 16..31 (K=64,     N=128: 2 kT x 8 nT)
//   L3: frags 32..47 (K=128,    N=64 : 4 kT x 4 nT)
//   L4: frags 48..49 (K=64,     N=16pad: 2 kT x 1 nT)
#define NFRAG      50
#define FRAG_HALFS 512

__device__ __forceinline__ float gelu_exact(float x) {
  return 0.5f * x * (1.0f + erff(x * 0.70710678118654752f));
}
__device__ __forceinline__ float sigmoid_f(float x) {
  return 1.0f / (1.0f + __expf(-x));
}

// --------------------------- weight prep kernel ----------------------------
// Convert f32 row-major W[K,N] into f16 WMMA B-fragments (32x16 KxN tiles):
//   lane 0..15 -> col = nT*16 + lane, K chunks {kT*32+0..7, kT*32+16..23}
//   lane 16..31-> col = nT*16 + (lane-16), K chunks {kT*32+8..15, kT*32+24..31}
__global__ __launch_bounds__(256) void prep_weights_kernel(
    const float* __restrict__ W1, const float* __restrict__ W2,
    const float* __restrict__ W3, const float* __restrict__ W4,
    _Float16* __restrict__ ws) {
  int idx = blockIdx.x * 256 + threadIdx.x;
  if (idx >= NFRAG * FRAG_HALFS) return;
  int gfrag  = idx >> 9;       // /512
  int within = idx & 511;
  int lane   = within >> 4;
  int hi     = within & 15;

  const float* Wp; int Korig, Nout, nTiles, lfrag;
  if (gfrag < 16)      { Wp = W1; Korig = 108; Nout = 64;  nTiles = 4; lfrag = gfrag;      }
  else if (gfrag < 32) { Wp = W2; Korig = 64;  Nout = 128; nTiles = 8; lfrag = gfrag - 16; }
  else if (gfrag < 48) { Wp = W3; Korig = 128; Nout = 64;  nTiles = 4; lfrag = gfrag - 32; }
  else                 { Wp = W4; Korig = 64;  Nout = 4;   nTiles = 1; lfrag = gfrag - 48; }

  int kT  = lfrag / nTiles;
  int nT  = lfrag % nTiles;
  int col = nT * 16 + (lane & 15);
  int k   = kT * 32 + ((lane >> 4) << 3) + (hi & 7) + ((hi & 8) << 1);

  float v = 0.0f;
  if (k < Korig && col < Nout) v = Wp[k * Nout + col];
  ws[gfrag * FRAG_HALFS + lane * 16 + hi] = (_Float16)v;
}

// --------------------------- fragment loaders ------------------------------
// A fragment (16x32 f16) from LDS staging tile [16][Kpad] row-major halfs.
//   lane 0..15 : row=lane,   K chunks {kT*32+0..7, kT*32+16..23}
//   lane 16..31: row=lane-16,K chunks {kT*32+8..15, kT*32+24..31}
__device__ __forceinline__ v16h load_a_frag(const _Float16* stg, int Kpad,
                                            int kT, int lane) {
  int row = lane & 15;
  int kb  = kT * 32 + ((lane >> 4) << 3);
  const _Float16* p = stg + row * Kpad + kb;
  v8h lo = *(const v8h*)(p);        // 16B aligned
  v8h hi = *(const v8h*)(p + 16);   // 16B aligned
  return __builtin_shufflevector(lo, hi, 0, 1, 2, 3, 4, 5, 6, 7,
                                 8, 9, 10, 11, 12, 13, 14, 15);
}

__device__ __forceinline__ v16h load_b_frag(const _Float16* __restrict__ ws,
                                            int gfrag, int lane) {
  return *(const v16h*)(ws + gfrag * FRAG_HALFS + lane * 16);  // 32B/lane
}

// ------------------------------ main kernel --------------------------------
__global__ __launch_bounds__(256) void vfx_patch_decoder_kernel(
    const float* __restrict__ latent, const int* __restrict__ raw_pos,
    const float* __restrict__ control,
    const float* __restrict__ b1, const float* __restrict__ b2,
    const float* __restrict__ b3, const float* __restrict__ b4,
    const _Float16* __restrict__ ws, float* __restrict__ out) {
  __shared__ _Float16 lds[8 * 2048];   // 8 waves x [16][128] halfs = 32 KB

  const int lane  = threadIdx.x & 31;
  const int wave  = threadIdx.x >> 5;
  _Float16* stg   = lds + wave * 2048;
  const int row   = lane & 15;
  const int rBase = (lane < 16) ? 0 : 8;   // C-tile row base for this lane
  const int colLo = lane & 15;             // C-tile column within nT
  const int pixelBase = (blockIdx.x * 8 + wave) * 16;
  const int p = pixelBase + row;

  // ---- build 16x108 patch (padded to 16x128) in LDS as f16 ----
  const int2   rp = ((const int2*)raw_pos)[p];
  const float2 ct = ((const float2*)control)[p];
  const _Float16 cxh = (_Float16)ct.x, cyh = (_Float16)ct.y;

  if (lane < 16) {  // zero-pad K = 108..127 for this row
    v4h z = {};
    _Float16* zp = stg + row * 128 + 108;
#pragma unroll
    for (int i = 0; i < 5; ++i) *(v4h*)(zp + i * 4) = z;
  }

  const float4* lat4 = (const float4*)latent;
  const int jStart = (lane < 16) ? 0 : 5;
  const int jEnd   = (lane < 16) ? 5 : 9;
#pragma unroll
  for (int j = 0; j < 9; ++j) {
    if (j < jStart || j >= jEnd) continue;
    int dx = j / 3 - 1, dy = j % 3 - 1;
    int ex = min(max(rp.x + dx, 0), HW - 1);
    int ey = min(max(rp.y + dy, 0), HW - 1);
    float4 lv = lat4[ey * HW + ex];           // random 16B gather, L2-resident
    float xf = (float)ex * (1.0f / HW);
    float yf = (float)ey * (1.0f / HW);
    float s1x = __sinf(6.28318530717958647f * xf);
    float s1y = __sinf(6.28318530717958647f * yf);
    float s2x = __sinf(12.5663706143591729f * xf);
    float s2y = __sinf(12.5663706143591729f * yf);
    _Float16* dst = stg + row * 128 + j * 12;   // 8B aligned (j*24 bytes)
    *(v4h*)(dst)     = (v4h){(_Float16)lv.x, (_Float16)lv.y,
                             (_Float16)lv.z, (_Float16)lv.w};
    *(v4h*)(dst + 4) = (v4h){(_Float16)xf, (_Float16)yf,
                             (_Float16)s1x, (_Float16)s1y};
    *(v4h*)(dst + 8) = (v4h){(_Float16)s2x, (_Float16)s2y, cxh, cyh};
  }
  asm volatile("s_wait_dscnt 0" ::: "memory");

  // ---- layer 1: [16,108pad128] x [108,64] + b1, GELU ----
  v8f c1[4];
#pragma unroll
  for (int nT = 0; nT < 4; ++nT) {
    float bv = b1[nT * 16 + colLo];
#pragma unroll
    for (int r = 0; r < 8; ++r) c1[nT][r] = bv;
  }
#pragma unroll
  for (int kT = 0; kT < 4; ++kT) {
    v16h a = load_a_frag(stg, 128, kT, lane);
#pragma unroll
    for (int nT = 0; nT < 4; ++nT) {
      v16h b = load_b_frag(ws, 0 + kT * 4 + nT, lane);
      c1[nT] = __builtin_amdgcn_wmma_f32_16x16x32_f16(
          false, a, false, b, (short)0, c1[nT], false, false);
    }
  }
#pragma unroll
  for (int nT = 0; nT < 4; ++nT)
#pragma unroll
    for (int r = 0; r < 8; ++r)
      stg[(rBase + r) * 64 + nT * 16 + colLo] = (_Float16)gelu_exact(c1[nT][r]);
  asm volatile("s_wait_dscnt 0" ::: "memory");

  // ---- layer 2: [16,64] x [64,128] + b2, GELU ----
  v8f c2[8];
#pragma unroll
  for (int nT = 0; nT < 8; ++nT) {
    float bv = b2[nT * 16 + colLo];
#pragma unroll
    for (int r = 0; r < 8; ++r) c2[nT][r] = bv;
  }
#pragma unroll
  for (int kT = 0; kT < 2; ++kT) {
    v16h a = load_a_frag(stg, 64, kT, lane);
#pragma unroll
    for (int nT = 0; nT < 8; ++nT) {
      v16h b = load_b_frag(ws, 16 + kT * 8 + nT, lane);
      c2[nT] = __builtin_amdgcn_wmma_f32_16x16x32_f16(
          false, a, false, b, (short)0, c2[nT], false, false);
    }
  }
#pragma unroll
  for (int nT = 0; nT < 8; ++nT)
#pragma unroll
    for (int r = 0; r < 8; ++r)
      stg[(rBase + r) * 128 + nT * 16 + colLo] = (_Float16)gelu_exact(c2[nT][r]);
  asm volatile("s_wait_dscnt 0" ::: "memory");

  // ---- layer 3: [16,128] x [128,64] + b3, GELU ----
  v8f c3[4];
#pragma unroll
  for (int nT = 0; nT < 4; ++nT) {
    float bv = b3[nT * 16 + colLo];
#pragma unroll
    for (int r = 0; r < 8; ++r) c3[nT][r] = bv;
  }
#pragma unroll
  for (int kT = 0; kT < 4; ++kT) {
    v16h a = load_a_frag(stg, 128, kT, lane);
#pragma unroll
    for (int nT = 0; nT < 4; ++nT) {
      v16h b = load_b_frag(ws, 32 + kT * 4 + nT, lane);
      c3[nT] = __builtin_amdgcn_wmma_f32_16x16x32_f16(
          false, a, false, b, (short)0, c3[nT], false, false);
    }
  }
#pragma unroll
  for (int nT = 0; nT < 4; ++nT)
#pragma unroll
    for (int r = 0; r < 8; ++r)
      stg[(rBase + r) * 64 + nT * 16 + colLo] = (_Float16)gelu_exact(c3[nT][r]);
  asm volatile("s_wait_dscnt 0" ::: "memory");

  // ---- layer 4: [16,64] x [64,4pad16] + b4, sigmoid ----
  v8f c4;
  {
    float bv = (colLo < 4) ? b4[colLo] : 0.0f;
#pragma unroll
    for (int r = 0; r < 8; ++r) c4[r] = bv;
  }
#pragma unroll
  for (int kT = 0; kT < 2; ++kT) {
    v16h a = load_a_frag(stg, 64, kT, lane);
    v16h b = load_b_frag(ws, 48 + kT, lane);
    c4 = __builtin_amdgcn_wmma_f32_16x16x32_f16(
        false, a, false, b, (short)0, c4, false, false);
  }
  if (colLo < 4) {
#pragma unroll
    for (int r = 0; r < 8; ++r)
      out[(pixelBase + rBase + r) * 4 + colLo] = sigmoid_f(c4[r]);
  }
}

// ------------------------------- launcher ----------------------------------
extern "C" void kernel_launch(void* const* d_in, const int* in_sizes, int n_in,
                              void* d_out, int out_size, void* d_ws,
                              size_t ws_size, hipStream_t stream) {
  const float* latent  = (const float*)d_in[0];
  const int*   raw_pos = (const int*)d_in[1];
  const float* control = (const float*)d_in[2];
  const float* W1 = (const float*)d_in[3];
  const float* b1 = (const float*)d_in[4];
  const float* W2 = (const float*)d_in[5];
  const float* b2 = (const float*)d_in[6];
  const float* W3 = (const float*)d_in[7];
  const float* b3 = (const float*)d_in[8];
  const float* W4 = (const float*)d_in[9];
  const float* b4 = (const float*)d_in[10];
  _Float16* ws = (_Float16*)d_ws;   // needs 50*512*2 = 51200 bytes

  (void)in_sizes; (void)n_in; (void)out_size; (void)ws_size;

  int prepThreads = NFRAG * FRAG_HALFS;
  prep_weights_kernel<<<(prepThreads + 255) / 256, 256, 0, stream>>>(
      W1, W2, W3, W4, ws);

  // 16 pixels per wave, 8 waves per block -> 128 pixels/block
  vfx_patch_decoder_kernel<<<NPIX / 128, 256, 0, stream>>>(
      latent, raw_pos, control, b1, b2, b3, b4, ws, (float*)d_out);
}
